// KernelizedAttentionRegressor_71528385347628
// MI455X (gfx1250) — matrix operand order repaired and checked by reading, not compiled, not run on hardware
//
#include <hip/hip_runtime.h>
#include <hip/hip_bf16.h>

#define TT 512
#define DQ 8
#define FF 2048
#define BM 32
#define FCHUNK 128
#define NATTN 4
#define NMLP 2
#define BB 8192

typedef __bf16 v16bf __attribute__((ext_vector_type(16)));
typedef float  v8f   __attribute__((ext_vector_type(8)));

union Frag { unsigned int u[8]; v16bf v; };

__device__ __forceinline__ unsigned short f2bf(float f) {
  unsigned int u = __float_as_uint(f);
  u += 0x7FFFu + ((u >> 16) & 1u);
  return (unsigned short)(u >> 16);
}

__device__ __forceinline__ float silu_f(float v) {
  return v * (1.0f / (1.0f + __expf(-v)));
}

__device__ __forceinline__ v8f vzero() {
  v8f z = {0.f, 0.f, 0.f, 0.f, 0.f, 0.f, 0.f, 0.f};
  return z;
}

__device__ __forceinline__ v8f wmma_bf16(const Frag& a, const Frag& b, v8f c) {
  return __builtin_amdgcn_wmma_f32_16x16x32_bf16(
      /*neg_a=*/false, a.v, /*neg_b=*/false, b.v,
      /*c_mod=*/(short)0, c, /*reuse_a=*/false, /*reuse_b=*/false);
}

// A fragment (16x32 bf16, M x K) from LDS row-major [stride] ushort array.
// ISA layout: lanes 0-15 M=0..15; VGPR v<4: K = 2v,2v+1 (+8 for lanes16-31),
// VGPR v>=4: K = 16+2(v-4),.. (+8 for lanes 16-31).
__device__ __forceinline__ void load_a(const unsigned short* s, int stride,
                                       int mbase, int kbase, int lane, Frag& f) {
  int m  = mbase + (lane & 15);
  int k0 = kbase + ((lane & 16) ? 8 : 0);
  const uint4* p = (const uint4*)(s + (long)m * stride + k0);
  uint4 q0 = p[0];   // K = k0 .. k0+7
  uint4 q1 = p[2];   // K = k0+16 .. k0+23
  f.u[0] = q0.x; f.u[1] = q0.y; f.u[2] = q0.z; f.u[3] = q0.w;
  f.u[4] = q1.x; f.u[5] = q1.y; f.u[6] = q1.z; f.u[7] = q1.w;
}

// B fragment (32x16 bf16, K x N) from pre-swizzled tile: 256 dwords [lane][v].
__device__ __forceinline__ void load_b(const unsigned int* tile, int lane, Frag& f) {
  const uint4* p = (const uint4*)(tile + lane * 8);
  uint4 w0 = p[0], w1 = p[1];
  f.u[0] = w0.x; f.u[1] = w0.y; f.u[2] = w0.z; f.u[3] = w0.w;
  f.u[4] = w1.x; f.u[5] = w1.y; f.u[6] = w1.z; f.u[7] = w1.w;
}

// ---------------- prep: f32 [N][K] weights -> bf16 WMMA-B fragment tiles ----
// Tile (kt,nt) covers K=kt*32..+31, N=nt*16..+15; word (lane,v) packs
// W[n][k],W[n][k+1] with n = nt*16+(lane&15), k = kt*32+((lane&16)?8:0)+2v+(v>=4?8:0).
extern "C" __global__ void kar_prep(const float* __restrict__ src,
                                    unsigned int* __restrict__ dst,
                                    int N, int K, long total) {
  long wi = (long)blockIdx.x * blockDim.x + threadIdx.x;
  if (wi >= total) return;
  long perMat = (long)N * K / 2;
  int  mat  = (int)(wi / perMat);
  long r    = wi - (long)mat * perMat;
  int  tile = (int)(r >> 8);
  int  lw   = (int)(r & 255);
  int  lane = lw >> 3, v = lw & 7;
  int  ntiles = N >> 4;
  int  kt = tile / ntiles, nt = tile - kt * ntiles;
  int  n  = nt * 16 + (lane & 15);
  int  k  = kt * 32 + ((lane & 16) ? 8 : 0) + 2 * v + ((v >= 4) ? 8 : 0);
  const float* m = src + (long)mat * N * K + (long)n * K + k;
  unsigned int lo = f2bf(m[0]);
  unsigned int hi = f2bf(m[1]);
  dst[wi] = lo | (hi << 16);
}

// ---------------- main fused kernel ----------------------------------------
struct KarParams {
  const float *x, *omega, *G, *phase, *l1_w, *l1_b;
  const float *mu_b, *sg_b;
  const float *aln_q_g, *aln_q_b, *aln_f_g, *aln_f_b;
  const float *a_b1, *a_b2;
  const float *mln_g, *mln_b, *m_b1, *m_b2;
  const float *ro_w, *ro_b;
  const unsigned int *mu_f, *sg_f, *aw1_f, *aw2_f, *mw1_f, *mw2_f;
  float* out;
};

__device__ __forceinline__ void layer_norm(const float* sQ, unsigned short* sh,
                                           const float* g, const float* bt, int tid) {
  int r = tid >> 3, sub = tid & 7;        // 8 lanes per row, contiguous in wave
  const float* row = sQ + r * TT;
  float s = 0.f, ss = 0.f;
  int c0 = sub * 64;
  for (int c = c0; c < c0 + 64; ++c) { float q = row[c]; s += q; ss += q * q; }
#pragma unroll
  for (int off = 1; off < 8; off <<= 1) {
    s  += __shfl_xor(s,  off, 32);
    ss += __shfl_xor(ss, off, 32);
  }
  float mean = s * (1.f / TT);
  float var  = ss * (1.f / TT) - mean * mean;
  float rs   = rsqrtf(var + 1e-5f);
  for (int c = c0; c < c0 + 64; ++c) {
    float v = (row[c] - mean) * rs * g[c] + bt[c];
    sh[r * TT + c] = f2bf(v);
  }
}

// [32,512] output, K=512. Wave w: M-tile (w&1), N-tiles (w>>1)*8 .. +7.
__device__ __forceinline__ void gemm_32x512(const unsigned short* A,
                                            const unsigned int* Wf,
                                            int wave, int lane, v8f acc[8]) {
  const int mt  = (wave & 1) * 16;
  const int ntb = (wave >> 1) * 8;
  for (int kt = 0; kt < 16; ++kt) {
    Frag a; load_a(A, TT, mt, kt * 32, lane, a);
    const unsigned int* base = Wf + (long)(kt * (TT / 16) + ntb) * 256;
#pragma unroll
    for (int j = 0; j < 8; ++j) {
      Frag b; load_b(base + j * 256, lane, b);
      acc[j] = wmma_bf16(a, b, acc[j]);
    }
  }
}

__device__ __forceinline__ void ffn_block(float* sQ, unsigned short* sh,
                                          unsigned short* sU,
                                          const float* g, const float* bt,
                                          const unsigned int* w1f, const float* b1,
                                          const unsigned int* w2f, const float* b2,
                                          int tid) {
  __syncthreads();
  layer_norm(sQ, sh, g, bt, tid);
  __syncthreads();
  const int wave = tid >> 5, lane = tid & 31;
  const int mt   = (wave & 1) * 16;
  const int ntb  = (wave >> 1) * 8;     // output N-tiles (512 cols)
  const int nt1b = (wave >> 1) * 2;     // chunk N-tiles (128 cols)
  v8f qacc[8];
#pragma unroll
  for (int j = 0; j < 8; ++j) qacc[j] = vzero();

  for (int fc = 0; fc < FF / FCHUNK; ++fc) {
    // GEMM1: u = h @ w1^T for this 128-col chunk
    v8f uacc[2] = { vzero(), vzero() };
    for (int kt = 0; kt < 16; ++kt) {
      Frag a; load_a(sh, TT, mt, kt * 32, lane, a);
#pragma unroll
      for (int j = 0; j < 2; ++j) {
        int ntg = fc * 8 + nt1b + j;    // tile index within N=2048
        Frag b; load_b(w1f + (long)(kt * (FF / 16) + ntg) * 256, lane, b);
        uacc[j] = wmma_bf16(a, b, uacc[j]);
      }
    }
    // bias + silu -> bf16 chunk in LDS
#pragma unroll
    for (int j = 0; j < 2; ++j) {
      int fl = (nt1b + j) * 16 + (lane & 15);
      float bb = b1[fc * FCHUNK + fl];
#pragma unroll
      for (int v = 0; v < 8; ++v) {
        int m = mt + v + ((lane & 16) ? 8 : 0);
        sU[m * FCHUNK + fl] = f2bf(silu_f(uacc[j][v] + bb));
      }
    }
    __syncthreads();
    // GEMM2: qacc += u_chunk @ w2^T (K=128)
#pragma unroll
    for (int kc = 0; kc < 4; ++kc) {
      Frag a; load_a(sU, FCHUNK, mt, kc * 32, lane, a);
      int ktg = fc * 4 + kc;            // K-tile within K=2048
#pragma unroll
      for (int j = 0; j < 8; ++j) {
        Frag b; load_b(w2f + (long)(ktg * (TT / 16) + ntb + j) * 256, lane, b);
        qacc[j] = wmma_bf16(a, b, qacc[j]);
      }
    }
    __syncthreads();
  }
  // residual + bias
#pragma unroll
  for (int j = 0; j < 8; ++j) {
    int n = (ntb + j) * 16 + (lane & 15);
    float bb = b2[n];
#pragma unroll
    for (int v = 0; v < 8; ++v) {
      int m = mt + v + ((lane & 16) ? 8 : 0);
      sQ[m * TT + n] += qacc[j][v] + bb;
    }
  }
  __syncthreads();
}

extern "C" __global__ __launch_bounds__(256, 1)
void kar_main(KarParams P) {
  extern __shared__ char smem[];
  float4*         sOGP = (float4*)smem;                     // [TT*DQ] (o0,o1,G,phase) 64KB
  float*          sQ   = (float*)(sOGP + TT * DQ);          // [BM*TT] 64KB
  unsigned short* sh   = (unsigned short*)(sQ + BM * TT);   // [BM*TT] 32KB
  unsigned short* sU   = (unsigned short*)(sh + BM * TT);   // [BM*FCHUNK] 8KB
  float*          sx   = (float*)(sU + BM * FCHUNK);        // [BM*2]

  const int tid  = threadIdx.x;
  const int wave = tid >> 5, lane = tid & 31;
  const int row0 = blockIdx.x * BM;

  if (tid < BM * 2) sx[tid] = P.x[row0 * 2 + tid];
  for (int idx = tid; idx < TT * DQ; idx += 256)
    sOGP[idx] = make_float4(P.omega[idx * 2], P.omega[idx * 2 + 1],
                            P.G[idx], P.phase[idx]);
  __syncthreads();

  // Q0 = silu(x @ l1_w^T + l1_b)
  for (int idx = tid; idx < BM * TT; idx += 256) {
    int m = idx >> 9, n = idx & (TT - 1);
    float q = sx[m * 2] * P.l1_w[n * 2] + sx[m * 2 + 1] * P.l1_w[n * 2 + 1] + P.l1_b[n];
    sQ[idx] = silu_f(q);
  }

  const int mt  = (wave & 1) * 16;
  const int ntb = (wave >> 1) * 8;

  for (int i = 0; i < NATTN; ++i) {
    __syncthreads();
    layer_norm(sQ, sh, P.aln_q_g + i * TT, P.aln_q_b + i * TT, tid);
    __syncthreads();

    v8f muacc[8], sgacc[8];
#pragma unroll
    for (int j = 0; j < 8; ++j) { muacc[j] = vzero(); sgacc[j] = vzero(); }
    gemm_32x512(sh, P.mu_f + (long)i * (TT * TT / 2), wave, lane, muacc);
    gemm_32x512(sh, P.sg_f + (long)i * (TT * TT / 2), wave, lane, sgacc);

    // attention epilogue: Q += sum_k exp(-(G-mu)^2/(2 sig^2+eps)) * cos(omega.x+phase)
#pragma unroll
    for (int j = 0; j < 8; ++j) {
      int n = (ntb + j) * 16 + (lane & 15);
      float mb = P.mu_b[i * TT + n], sb = P.sg_b[i * TT + n];
#pragma unroll
      for (int v = 0; v < 8; ++v) {
        int m = mt + v + ((lane & 16) ? 8 : 0);
        float mu  = tanhf(muacc[j][v] + mb);
        float sg  = sgacc[j][v] + sb;
        float inv = -0.5f / (sg * sg + 1e-8f);
        float x0 = sx[m * 2], x1 = sx[m * 2 + 1];
        float ssum = 0.f;
#pragma unroll
        for (int k = 0; k < DQ; ++k) {
          float4 t = sOGP[n * DQ + k];
          float hk = __cosf(t.x * x0 + t.y * x1 + t.w);
          float d  = t.z - mu;
          ssum += __expf(inv * d * d) * hk;
        }
        sQ[m * TT + n] += ssum;
      }
    }

    ffn_block(sQ, sh, sU, P.aln_f_g + i * TT, P.aln_f_b + i * TT,
              P.aw1_f + (long)i * (FF * TT / 2), P.a_b1 + i * FF,
              P.aw2_f + (long)i * (FF * TT / 2), P.a_b2 + i * TT, tid);
  }

  for (int i = 0; i < NMLP; ++i) {
    ffn_block(sQ, sh, sU, P.mln_g + i * TT, P.mln_b + i * TT,
              P.mw1_f + (long)i * (FF * TT / 2), P.m_b1 + i * FF,
              P.mw2_f + (long)i * (FF * TT / 2), P.m_b2 + i * TT, tid);
  }

  __syncthreads();
  // out = silu(Q) @ ro_w^T + ro_b
  {
    int r = tid >> 3, sub = tid & 7;
    float s = 0.f;
    int c0 = sub * 64;
    for (int c = c0; c < c0 + 64; ++c) {
      float q = sQ[r * TT + c];
      s += silu_f(q) * P.ro_w[c];
    }
#pragma unroll
    for (int off = 1; off < 8; off <<= 1) s += __shfl_xor(s, off, 32);
    if (sub == 0) P.out[row0 + r] = s + P.ro_b[0];
  }
}

extern "C" void kernel_launch(void* const* d_in, const int* in_sizes, int n_in,
                              void* d_out, int out_size, void* d_ws, size_t ws_size,
                              hipStream_t stream) {
  const float* x       = (const float*)d_in[0];
  const float* omega   = (const float*)d_in[1];
  const float* G       = (const float*)d_in[2];
  const float* phase   = (const float*)d_in[3];
  const float* l1_w    = (const float*)d_in[4];
  const float* l1_b    = (const float*)d_in[5];
  const float* mu_w    = (const float*)d_in[6];
  const float* mu_b    = (const float*)d_in[7];
  const float* sg_w    = (const float*)d_in[8];
  const float* sg_b    = (const float*)d_in[9];
  const float* aln_q_g = (const float*)d_in[10];
  const float* aln_q_b = (const float*)d_in[11];
  const float* aln_f_g = (const float*)d_in[12];
  const float* aln_f_b = (const float*)d_in[13];
  const float* a_w1    = (const float*)d_in[14];
  const float* a_b1    = (const float*)d_in[15];
  const float* a_w2    = (const float*)d_in[16];
  const float* a_b2    = (const float*)d_in[17];
  const float* mln_g   = (const float*)d_in[18];
  const float* mln_b   = (const float*)d_in[19];
  const float* m_w1    = (const float*)d_in[20];
  const float* m_b1    = (const float*)d_in[21];
  const float* m_w2    = (const float*)d_in[22];
  const float* m_b2    = (const float*)d_in[23];
  const float* ro_w    = (const float*)d_in[24];
  const float* ro_b    = (const float*)d_in[25];

  unsigned int* ws = (unsigned int*)d_ws;
  const long WM  = (long)TT * TT / 2;   // dwords per 512x512 matrix (bf16 packed)
  const long W1M = (long)FF * TT / 2;   // dwords per 2048x512 / 512x2048 matrix
  unsigned int* mu_f  = ws;
  unsigned int* sg_f  = mu_f  + 4 * WM;
  unsigned int* aw1_f = sg_f  + 4 * WM;
  unsigned int* aw2_f = aw1_f + 4 * W1M;
  unsigned int* mw1_f = aw2_f + 4 * W1M;
  unsigned int* mw2_f = mw1_f + 2 * W1M;

  auto prep = [&](const float* src, unsigned int* dst, int N, int K, int nmat) {
    long total = (long)nmat * N * K / 2;
    int blocks = (int)((total + 255) / 256);
    hipLaunchKernelGGL(kar_prep, dim3(blocks), dim3(256), 0, stream,
                       src, dst, N, K, total);
  };
  prep(mu_w, mu_f, TT, TT, 4);
  prep(sg_w, sg_f, TT, TT, 4);
  prep(a_w1, aw1_f, FF, TT, 4);
  prep(a_w2, aw2_f, TT, FF, 4);
  prep(m_w1, mw1_f, FF, TT, 2);
  prep(m_w2, mw2_f, TT, FF, 2);

  KarParams P;
  P.x = x; P.omega = omega; P.G = G; P.phase = phase; P.l1_w = l1_w; P.l1_b = l1_b;
  P.mu_b = mu_b; P.sg_b = sg_b;
  P.aln_q_g = aln_q_g; P.aln_q_b = aln_q_b; P.aln_f_g = aln_f_g; P.aln_f_b = aln_f_b;
  P.a_b1 = a_b1; P.a_b2 = a_b2;
  P.mln_g = mln_g; P.mln_b = mln_b; P.m_b1 = m_b1; P.m_b2 = m_b2;
  P.ro_w = ro_w; P.ro_b = ro_b;
  P.mu_f = mu_f; P.sg_f = sg_f; P.aw1_f = aw1_f; P.aw2_f = aw2_f;
  P.mw1_f = mw1_f; P.mw2_f = mw2_f;
  P.out = (float*)d_out;

  size_t shmem = (size_t)TT * DQ * 16      // sOGP
               + (size_t)BM * TT * 4       // sQ
               + (size_t)BM * TT * 2       // sh
               + (size_t)BM * FCHUNK * 2   // sU
               + (size_t)BM * 2 * 4;       // sx   = 172,288 bytes
  hipLaunchKernelGGL(kar_main, dim3(BB / BM), dim3(256), shmem, stream, P);
}
